// Mozart_22393959481574
// MI455X (gfx1250) — matrix operand order repaired and checked by reading, not compiled, not run on hardware
//
#include <hip/hip_runtime.h>
#include <hip/hip_bf16.h>

typedef __attribute__((ext_vector_type(16))) __bf16 v16bf;
typedef __attribute__((ext_vector_type(8)))  __bf16 v8bf;
typedef __attribute__((ext_vector_type(8)))  float  v8f;
typedef __attribute__((ext_vector_type(4)))  float  v4f;
typedef __attribute__((ext_vector_type(4)))  unsigned int u32x4;
typedef __attribute__((ext_vector_type(8)))  int    i32x8;
typedef __attribute__((ext_vector_type(4)))  int    i32x4;

#define WMMA_BF16(a, b, c) \
  __builtin_amdgcn_wmma_f32_16x16x32_bf16(false, (a), false, (b), (short)0, (c), false, false)

// Problem constants
#define CB  4
#define CT  2048
#define CE  512
#define CH  8
#define CL  5
#define CV  130
#define CBT 8192          // B*T
#define CF  2048          // 4*E

// ---------------------------------------------------------------------------
// TDM: DMA a [rows x 32] bf16 tile (row stride K elements) from global into
// LDS at lds_off, padded to a 40-element (80B) LDS row stride.
// D# per CDNA5 ISA 8.3/8.4: group0 = {count=1, lds_addr, global_addr, type=2},
// group1 = {data_size=2B, pad_enable, pad_interval=16DW, pad_amount=4DW,
//           tensor_dim0=K, tensor_dim1=rows, tile=32 x rows, stride0=K}.
// This toolchain's builtin takes the 6-arg form (groups 2/3 + trailing i32x8
// unused for 2-D tiles, cpol=0).
// ---------------------------------------------------------------------------
__device__ __forceinline__ void tdm_tile(const __bf16* g, unsigned lds_off,
                                         int K, int rows) {
  unsigned long long ga = (unsigned long long)(uintptr_t)g;
  u32x4 g0;
  g0[0] = 1u;                                   // count=1 (valid descriptor)
  g0[1] = lds_off;                              // LDS byte address
  g0[2] = (unsigned)ga;                         // global addr [31:0]
  g0[3] = (unsigned)(ga >> 32) | 0x80000000u;   // addr [56:32] | type=2
  i32x8 g1;
  g1[0] = (int)0x06D10000u;   // data_size=1(2B) | pad_en | interval=3(16DW) | amount=3(4DW)
  g1[1] = (int)((unsigned)K << 16);     // tensor_dim0[15:0] in bits[63:48]
  g1[2] = (int)((unsigned)rows << 16);  // tensor_dim0 hi=0 | tensor_dim1[15:0]
  g1[3] = (int)(32u << 16);             // tensor_dim1 hi=0 | tile_dim0=32
  g1[4] = rows;                         // tile_dim1 | tile_dim2=0
  g1[5] = K;                            // tensor_dim0_stride [31:0]
  g1[6] = 0;
  g1[7] = 0;
  i32x4 z4 = {0, 0, 0, 0};
  i32x8 z8 = {0, 0, 0, 0, 0, 0, 0, 0};
  __builtin_amdgcn_tensor_load_to_lds(g0, g1, z4, z4, z8, 0);
}

// ---------------------------------------------------------------------------
// Embedding: y[bt, :] = emb[tokens[bt], :] + pos[t, :]
// ---------------------------------------------------------------------------
__global__ __launch_bounds__(256) void k_embed(const int* __restrict__ tok,
                                               const float* __restrict__ emb,
                                               const float* __restrict__ pos,
                                               float* __restrict__ y) {
  int i = blockIdx.x * 256 + threadIdx.x;   // over BT*E/4 float4s
  int bt = i >> 7;                          // 128 float4 per token
  int c4 = (i & 127) * 4;
  int t  = bt & (CT - 1);
  int tk = tok[bt];
  v4f e = *(const v4f*)(emb + (size_t)tk * CE + c4);
  v4f p = *(const v4f*)(pos + (size_t)t * CE + c4);
  *(v4f*)(y + (size_t)bt * CE + c4) = e + p;
}

// ---------------------------------------------------------------------------
// f32 -> bf16 conversion
// ---------------------------------------------------------------------------
__global__ __launch_bounds__(256) void k_cvt(const float* __restrict__ src,
                                             __bf16* __restrict__ dst, int n) {
  int base = (blockIdx.x * 256 + threadIdx.x) * 4;
  if (base >= n) return;
  v4f v = *(const v4f*)(src + base);
  dst[base + 0] = (__bf16)v[0];
  dst[base + 1] = (__bf16)v[1];
  dst[base + 2] = (__bf16)v[2];
  dst[base + 3] = (__bf16)v[3];
}

// ---------------------------------------------------------------------------
// LayerNorm over E=512: one wave per token, 8 tokens per block.
// ---------------------------------------------------------------------------
__global__ __launch_bounds__(256) void k_ln(const float* __restrict__ x,
                                            const float* __restrict__ g,
                                            const float* __restrict__ b,
                                            float* __restrict__ o32,
                                            __bf16* __restrict__ o16) {
  const int wave = threadIdx.x >> 5, lane = threadIdx.x & 31;
  const int tok = blockIdx.x * 8 + wave;
  const float* xr = x + (size_t)tok * CE;
  float vals[16];
  float s = 0.f, s2 = 0.f;
#pragma unroll
  for (int i = 0; i < 4; ++i) {
    v4f v = *(const v4f*)(xr + i * 128 + lane * 4);
#pragma unroll
    for (int j = 0; j < 4; ++j) {
      float f = v[j];
      vals[i * 4 + j] = f;
      s += f;
      s2 = fmaf(f, f, s2);
    }
  }
#pragma unroll
  for (int off = 16; off > 0; off >>= 1) {
    s  += __shfl_xor(s, off);
    s2 += __shfl_xor(s2, off);
  }
  float mu   = s * (1.f / CE);
  float var  = s2 * (1.f / CE) - mu * mu;
  float rstd = rsqrtf(var + 1e-5f);
#pragma unroll
  for (int i = 0; i < 4; ++i) {
#pragma unroll
    for (int j = 0; j < 4; ++j) {
      int col = i * 128 + lane * 4 + j;
      float o = (vals[i * 4 + j] - mu) * rstd * g[col] + b[col];
      o32[(size_t)tok * CE + col] = o;
      o16[(size_t)tok * CE + col] = (__bf16)o;
    }
  }
}

// ---------------------------------------------------------------------------
// WMMA GEMM: C[M,N] = act( A[M,K] @ W[N,K]^T + bias [+ addend] )
// Block tile 64x128x32, 8 waves (4x2), 4 WMMA/wave/kstep, one reused A-frag.
// A/W tiles DMA'd by the Tensor Data Mover, double-buffered in LDS: wave 0
// issues tensor_load_to_lds for slab k+1, all waves compute slab k, wave 0
// s_wait_tensorcnt(0) before the barrier. blockIdx.z: W += z*wz, C += z*cz.
// ---------------------------------------------------------------------------
template <bool RELU, bool TRANS_OUT, bool OUT_BF16, bool OUT_F32>
__global__ __launch_bounds__(256) void k_gemm_bt(
    const __bf16* __restrict__ A, const __bf16* __restrict__ W,
    const float* __restrict__ bias, const float* __restrict__ addend,
    __bf16* __restrict__ Cb, float* __restrict__ Cf,
    int M, int N, int K, size_t wz, size_t cz) {
  const int bm = blockIdx.x * 64;
  const int bn = blockIdx.y * 128;
  W += (size_t)blockIdx.z * wz;
  if (OUT_BF16) Cb += (size_t)blockIdx.z * cz;
  if (OUT_F32)  Cf += (size_t)blockIdx.z * cz;

  const int tid = threadIdx.x, lane = tid & 31, wave = tid >> 5;
  const int wm = wave >> 1, wn = wave & 1;
  const int m = lane & 15, hi = lane >> 4;
  const int wrows = (N - bn < 128) ? (N - bn) : 128;   // valid W rows this block

  __shared__ __align__(16) __bf16 sA[2][64 * 40];
  __shared__ __align__(16) __bf16 sB[2][128 * 40];

  // preload slab 0
  if (wave == 0) {
    tdm_tile(A + (size_t)bm * K, (unsigned)(uintptr_t)&sA[0][0], K, 64);
    tdm_tile(W + (size_t)bn * K, (unsigned)(uintptr_t)&sB[0][0], K, wrows);
    __builtin_amdgcn_s_wait_tensorcnt(0);
  }
  __syncthreads();

  v8f c0 = {}, c1 = {}, c2 = {}, c3 = {};
  const int NK = K >> 5;
  for (int kk = 0; kk < NK; ++kk) {
    const int cur = kk & 1;
    if (wave == 0 && kk + 1 < NK) {   // DMA next slab while computing this one
      tdm_tile(A + (size_t)bm * K + (kk + 1) * 32,
               (unsigned)(uintptr_t)&sA[cur ^ 1][0], K, 64);
      tdm_tile(W + (size_t)bn * K + (kk + 1) * 32,
               (unsigned)(uintptr_t)&sB[cur ^ 1][0], K, wrows);
    }
    const __bf16* la = &sA[cur][0];
    const __bf16* lb = &sB[cur][0];
    v16bf a, b0, b1, b2, b3;
    ((v8bf*)&a)[0]  = *(const v8bf*)&la[(wm * 16 + m) * 40 + hi * 16];
    ((v8bf*)&a)[1]  = *(const v8bf*)&la[(wm * 16 + m) * 40 + hi * 16 + 8];
    const int nb = wn * 64;
    ((v8bf*)&b0)[0] = *(const v8bf*)&lb[(nb + m) * 40 + hi * 16];
    ((v8bf*)&b0)[1] = *(const v8bf*)&lb[(nb + m) * 40 + hi * 16 + 8];
    ((v8bf*)&b1)[0] = *(const v8bf*)&lb[(nb + 16 + m) * 40 + hi * 16];
    ((v8bf*)&b1)[1] = *(const v8bf*)&lb[(nb + 16 + m) * 40 + hi * 16 + 8];
    ((v8bf*)&b2)[0] = *(const v8bf*)&lb[(nb + 32 + m) * 40 + hi * 16];
    ((v8bf*)&b2)[1] = *(const v8bf*)&lb[(nb + 32 + m) * 40 + hi * 16 + 8];
    ((v8bf*)&b3)[0] = *(const v8bf*)&lb[(nb + 48 + m) * 40 + hi * 16];
    ((v8bf*)&b3)[1] = *(const v8bf*)&lb[(nb + 48 + m) * 40 + hi * 16 + 8];
    c0 = WMMA_BF16(a, b0, c0);
    c1 = WMMA_BF16(a, b1, c1);
    c2 = WMMA_BF16(a, b2, c2);
    c3 = WMMA_BF16(a, b3, c3);
    if (wave == 0 && kk + 1 < NK) __builtin_amdgcn_s_wait_tensorcnt(0);
    __syncthreads();
  }

#pragma unroll
  for (int j = 0; j < 8; ++j) {
    int row = bm + wm * 16 + j + hi * 8;
#pragma unroll
    for (int t = 0; t < 4; ++t) {
      int col = bn + wn * 64 + t * 16 + m;
      if (col >= N) continue;
      float v = (t == 0) ? c0[j] : (t == 1) ? c1[j] : (t == 2) ? c2[j] : c3[j];
      if (bias)   v += bias[col];
      if (RELU)   v = fmaxf(v, 0.f);
      if (addend) v += addend[(size_t)row * N + col];
      if (OUT_BF16) {
        if (TRANS_OUT) Cb[(size_t)col * M + row] = (__bf16)v;
        else           Cb[(size_t)row * N + col] = (__bf16)v;
      }
      if (OUT_F32) Cf[(size_t)row * N + col] = v;
    }
  }
}

// ---------------------------------------------------------------------------
// Flash attention, head_dim=512, causal, per-head output (summed later).
// Block: 8 waves; one (b,h,query-tile-of-16). Wave w owns D-slice [64w,64w+64).
// Q frags in registers; K / V^T B-frags batched straight from global; 16x32
// score tile reduced deterministically in LDS; online softmax in f32.
// q,k: [H, B*T, D] bf16.  vt: [H, D, B*T] bf16.  att: [H, B*T, D] f32.
// ---------------------------------------------------------------------------
__global__ __launch_bounds__(256) void k_flash(const __bf16* __restrict__ q,
                                               const __bf16* __restrict__ k,
                                               const __bf16* __restrict__ vt,
                                               float* __restrict__ att) {
  const int qt = blockIdx.x;
  const int bq = blockIdx.y;
  const int h  = blockIdx.z;
  const int tid = threadIdx.x, lane = tid & 31, wave = tid >> 5;
  const int m = lane & 15, hi = lane >> 4;
  const int dw = wave * 64;

  const size_t hbase = (size_t)h * CBT * CE;
  const __bf16* qh  = q  + hbase + (size_t)bq * CT * CE;
  const __bf16* kh  = k  + hbase + (size_t)bq * CT * CE;
  const __bf16* vth = vt + hbase + (size_t)bq * CT;       // row stride CBT
  float*        oh  = att + hbase + (size_t)bq * CT * CE;

  __shared__ float sSp[8][16 * 32];
  __shared__ float sS[16 * 32];
  __shared__ __align__(16) __bf16 sP[16 * 40];
  __shared__ float sM[16], sL[16], sAl[16];

  v16bf qa0, qa1;
  {
    const __bf16* qr = qh + (size_t)(qt * 16 + m) * CE + dw;
    ((v8bf*)&qa0)[0] = *(const v8bf*)(qr + hi * 16);
    ((v8bf*)&qa0)[1] = *(const v8bf*)(qr + hi * 16 + 8);
    ((v8bf*)&qa1)[0] = *(const v8bf*)(qr + 32 + hi * 16);
    ((v8bf*)&qa1)[1] = *(const v8bf*)(qr + 32 + hi * 16 + 8);
  }
  v8f o0 = {}, o1 = {}, o2 = {}, o3 = {};
  if (tid < 16) { sM[tid] = -3.0e38f; sL[tid] = 0.f; }
  __syncthreads();

  const int ktmax = (qt * 16 + 15) >> 5;
  for (int kt = 0; kt <= ktmax; ++kt) {
    // --- partial scores: batch all 4 K-frag loads, then 4 WMMAs ---
    v8f s0 = {}, s1 = {};
    {
      const __bf16* kr0 = kh + (size_t)(kt * 32 + m) * CE + dw;   // keys 0..15
      const __bf16* kr1 = kr0 + (size_t)16 * CE;                  // keys 16..31
      v16bf kb00, kb01, kb10, kb11;
      ((v8bf*)&kb00)[0] = *(const v8bf*)(kr0 + hi * 16);
      ((v8bf*)&kb00)[1] = *(const v8bf*)(kr0 + hi * 16 + 8);
      ((v8bf*)&kb01)[0] = *(const v8bf*)(kr0 + 32 + hi * 16);
      ((v8bf*)&kb01)[1] = *(const v8bf*)(kr0 + 32 + hi * 16 + 8);
      ((v8bf*)&kb10)[0] = *(const v8bf*)(kr1 + hi * 16);
      ((v8bf*)&kb10)[1] = *(const v8bf*)(kr1 + hi * 16 + 8);
      ((v8bf*)&kb11)[0] = *(const v8bf*)(kr1 + 32 + hi * 16);
      ((v8bf*)&kb11)[1] = *(const v8bf*)(kr1 + 32 + hi * 16 + 8);
      if (kt < ktmax)   // hint the next key tile toward the caches
        __builtin_prefetch(kr0 + 32 * CE, 0, 0);
      s0 = WMMA_BF16(qa0, kb00, s0);
      s0 = WMMA_BF16(qa1, kb01, s0);
      s1 = WMMA_BF16(qa0, kb10, s1);
      s1 = WMMA_BF16(qa1, kb11, s1);
    }
    float* sp = sSp[wave];
#pragma unroll
    for (int j = 0; j < 8; ++j) {
      sp[(j + hi * 8) * 32 + m]      = s0[j];
      sp[(j + hi * 8) * 32 + 16 + m] = s1[j];
    }
    __syncthreads();
#pragma unroll
    for (int e = tid; e < 512; e += 256) {    // deterministic cross-wave sum
      float acc = sSp[0][e];
#pragma unroll
      for (int w = 1; w < 8; ++w) acc += sSp[w][e];
      sS[e] = acc;
    }
    __syncthreads();

    // --- online softmax (one row per lane, tid<16) ---
    if (tid < 16) {
      const int qi = qt * 16 + tid;
      const float sc = 0.04419417382415922f;   // 512^-0.5
      float mx = sM[tid];
      float rmax = -3.0e38f;
#pragma unroll
      for (int c = 0; c < 32; ++c) {
        int ki = kt * 32 + c;
        float v = sS[tid * 32 + c] * sc;
        if (ki <= qi && v > rmax) rmax = v;
      }
      float mnew  = fmaxf(mx, rmax);
      float alpha = __expf(mx - mnew);
      float lsum  = 0.f;
#pragma unroll
      for (int c = 0; c < 32; ++c) {
        int ki = kt * 32 + c;
        float p = (ki <= qi) ? __expf(sS[tid * 32 + c] * sc - mnew) : 0.f;
        lsum += p;
        sP[tid * 40 + c] = (__bf16)p;
      }
      sL[tid]  = sL[tid] * alpha + lsum;
      sM[tid]  = mnew;
      sAl[tid] = alpha;
    }
    __syncthreads();

    // --- rescale O; batch 4 V-frag loads; 4 WMMAs ---
#pragma unroll
    for (int j = 0; j < 8; ++j) {
      float al = sAl[j + hi * 8];
      o0[j] *= al; o1[j] *= al; o2[j] *= al; o3[j] *= al;
    }
    v16bf pa;
    ((v8bf*)&pa)[0] = *(const v8bf*)&sP[m * 40 + hi * 16];
    ((v8bf*)&pa)[1] = *(const v8bf*)&sP[m * 40 + hi * 16 + 8];
    {
      v16bf vb0, vb1, vb2, vb3;
      const __bf16* vr0 = vth + (size_t)(dw + m) * CBT + kt * 32 + hi * 16;
      const __bf16* vr1 = vr0 + (size_t)16 * CBT;
      const __bf16* vr2 = vr0 + (size_t)32 * CBT;
      const __bf16* vr3 = vr0 + (size_t)48 * CBT;
      ((v8bf*)&vb0)[0] = *(const v8bf*)vr0;
      ((v8bf*)&vb0)[1] = *(const v8bf*)(vr0 + 8);
      ((v8bf*)&vb1)[0] = *(const v8bf*)vr1;
      ((v8bf*)&vb1)[1] = *(const v8bf*)(vr1 + 8);
      ((v8bf*)&vb2)[0] = *(const v8bf*)vr2;
      ((v8bf*)&vb2)[1] = *(const v8bf*)(vr2 + 8);
      ((v8bf*)&vb3)[0] = *(const v8bf*)vr3;
      ((v8bf*)&vb3)[1] = *(const v8bf*)(vr3 + 8);
      o0 = WMMA_BF16(pa, vb0, o0);
      o1 = WMMA_BF16(pa, vb1, o1);
      o2 = WMMA_BF16(pa, vb2, o2);
      o3 = WMMA_BF16(pa, vb3, o3);
    }
    __syncthreads();
  }

#pragma unroll
  for (int j = 0; j < 8; ++j) {
    int row = j + hi * 8;
    float inv = 1.0f / sL[row];
    float* op = oh + (size_t)(qt * 16 + row) * CE + dw + m;
    op[0]  = o0[j] * inv;
    op[16] = o1[j] * inv;
    op[32] = o2[j] * inv;
    op[48] = o3[j] * inv;
  }
}

// ---------------------------------------------------------------------------
// y += sum over heads of att
// ---------------------------------------------------------------------------
__global__ __launch_bounds__(256) void k_resid(const float* __restrict__ att,
                                               float* __restrict__ y) {
  int i = blockIdx.x * 256 + threadIdx.x;
  float acc = y[i];
#pragma unroll
  for (int h = 0; h < CH; ++h) acc += att[(size_t)h * (CBT * CE) + i];
  y[i] = acc;
}

// ---------------------------------------------------------------------------
// Host-side orchestration
// ---------------------------------------------------------------------------
extern "C" void kernel_launch(void* const* d_in, const int* in_sizes, int n_in,
                              void* d_out, int out_size, void* d_ws, size_t ws_size,
                              hipStream_t stream) {
  (void)in_sizes; (void)n_in; (void)out_size; (void)ws_size;
  const int*   tokens = (const int*)d_in[0];
  const float* emb    = (const float*)d_in[1];
  const float* pos    = (const float*)d_in[2];
  const float* ln1_g  = (const float*)d_in[3];
  const float* ln1_b  = (const float*)d_in[4];
  const float* Wq     = (const float*)d_in[5];
  const float* Wk     = (const float*)d_in[6];
  const float* Wv     = (const float*)d_in[7];
  const float* ln2_g  = (const float*)d_in[8];
  const float* ln2_b  = (const float*)d_in[9];
  const float* W1     = (const float*)d_in[10];
  const float* b1     = (const float*)d_in[11];
  const float* W2     = (const float*)d_in[12];
  const float* b2     = (const float*)d_in[13];
  const float* Wout   = (const float*)d_in[14];
  const float* bout   = (const float*)d_in[15];
  float* out = (float*)d_out;

  char* wp = (char*)d_ws;
  auto alloc = [&](size_t bytes) -> void* {
    void* p = wp;
    wp += (bytes + 255) & ~(size_t)255;
    return p;
  };
  float*   y   = (float*)  alloc((size_t)CBT * CE * 4);
  float*   h2  = (float*)  alloc((size_t)CBT * CE * 4);
  __bf16*  h16 = (__bf16*) alloc((size_t)CBT * CE * 2);
  __bf16*  qb  = (__bf16*) alloc((size_t)CH * CBT * CE * 2);
  __bf16*  kb  = (__bf16*) alloc((size_t)CH * CBT * CE * 2);
  __bf16*  vtb = (__bf16*) alloc((size_t)CH * CBT * CE * 2);
  float*   att = (float*)  alloc((size_t)CH * CBT * CE * 4);
  __bf16*  m1  = (__bf16*) alloc((size_t)CBT * CF * 2);
  __bf16*  wqb = (__bf16*) alloc((size_t)CL * CH * CE * CE * 2);
  __bf16*  wkb = (__bf16*) alloc((size_t)CL * CH * CE * CE * 2);
  __bf16*  wvb = (__bf16*) alloc((size_t)CL * CH * CE * CE * 2);
  __bf16*  w1b = (__bf16*) alloc((size_t)CL * CF * CE * 2);
  __bf16*  w2b = (__bf16*) alloc((size_t)CL * CE * CF * 2);
  __bf16*  wob = (__bf16*) alloc((size_t)CV * CE * 2);

  auto cvt = [&](const float* s, __bf16* d, size_t n) {
    k_cvt<<<dim3((unsigned)((n / 4 + 255) / 256)), 256, 0, stream>>>(s, d, (int)n);
  };
  cvt(Wq,   wqb, (size_t)CL * CH * CE * CE);
  cvt(Wk,   wkb, (size_t)CL * CH * CE * CE);
  cvt(Wv,   wvb, (size_t)CL * CH * CE * CE);
  cvt(W1,   w1b, (size_t)CL * CF * CE);
  cvt(W2,   w2b, (size_t)CL * CE * CF);
  cvt(Wout, wob, (size_t)CV * CE);

  k_embed<<<dim3(CBT * CE / 4 / 256), 256, 0, stream>>>(tokens, emb, pos, y);

  const size_t headW = (size_t)CE * CE;
  const size_t headC = (size_t)CBT * CE;

  for (int l = 0; l < CL; ++l) {
    const __bf16* wql = wqb + (size_t)l * CH * headW;
    const __bf16* wkl = wkb + (size_t)l * CH * headW;
    const __bf16* wvl = wvb + (size_t)l * CH * headW;
    const __bf16* w1l = w1b + (size_t)l * CF * CE;
    const __bf16* w2l = w2b + (size_t)l * CE * CF;

    k_ln<<<dim3(CBT / 8), 256, 0, stream>>>(y, ln1_g + l * CE, ln1_b + l * CE, h2, h16);

    k_gemm_bt<false, false, true, false><<<dim3(CBT / 64, CE / 128, CH), 256, 0, stream>>>(
        h16, wql, nullptr, nullptr, qb, nullptr, CBT, CE, CE, headW, headC);
    k_gemm_bt<false, false, true, false><<<dim3(CBT / 64, CE / 128, CH), 256, 0, stream>>>(
        h16, wkl, nullptr, nullptr, kb, nullptr, CBT, CE, CE, headW, headC);
    k_gemm_bt<false, true, true, false><<<dim3(CBT / 64, CE / 128, CH), 256, 0, stream>>>(
        h16, wvl, nullptr, nullptr, vtb, nullptr, CBT, CE, CE, headW, headC);

    k_flash<<<dim3(CT / 16, CB, CH), 256, 0, stream>>>(qb, kb, vtb, att);

    k_resid<<<dim3(CBT * CE / 256), 256, 0, stream>>>(att, y);

    k_ln<<<dim3(CBT / 8), 256, 0, stream>>>(y, ln2_g + l * CE, ln2_b + l * CE, h2, h16);

    k_gemm_bt<true, false, true, false><<<dim3(CBT / 64, CF / 128, 1), 256, 0, stream>>>(
        h16, w1l, b1 + (size_t)l * CF, nullptr, m1, nullptr, CBT, CF, CE, 0, 0);
    k_gemm_bt<false, false, false, true><<<dim3(CBT / 64, CE / 128, 1), 256, 0, stream>>>(
        m1, w2l, b2 + (size_t)l * CE, h2, nullptr, y, CBT, CE, CF, 0, 0);
  }

  cvt(y, h16, (size_t)CBT * CE);
  k_gemm_bt<false, false, false, true><<<dim3(CBT / 64, (CV + 127) / 128, 1), 256, 0, stream>>>(
      h16, wob, bout, nullptr, nullptr, out, CBT, CV, CE, 0, 0);
}
